// AtrousSelfAttention_43714177138969
// MI455X (gfx1250) — matrix operand order repaired and checked by reading, not compiled, not run on hardware
//
#include <hip/hip_runtime.h>
#include <hip/hip_bf16.h>
#include <math.h>

typedef __bf16 bf16;
typedef __attribute__((ext_vector_type(16))) __bf16 v16bf;
typedef __attribute__((ext_vector_type(8)))  float  v8f;

static constexpr int Bn = 4;
static constexpr int Cc = 256;     // Cin == Cout
static constexpr int Nn = 4096;    // H*W
static constexpr int HD = 128;     // head dim

#define WMMA_BF16(a, b, c) \
  __builtin_amdgcn_wmma_f32_16x16x32_bf16(false, (a), false, (b), (short)0, (c), false, false)

// A-operand channel order within each 32-block: linear (identity), so the
// A-fragment (k = 16*lanehalf + s) matches the contiguous B-fragment loads.
// It is ONE contiguous 32-byte load at offset lanehalf*16.
__host__ __device__ inline int swap34(int j) {
  return j;
}

// ---------------------------------------------------------------------------
// Prep: f32 -> bf16 conversions + layout changes
// ---------------------------------------------------------------------------
__global__ __launch_bounds__(256) void prep_x(const float* __restrict__ x,
                                              bf16* __restrict__ xt) {
  int idx = blockIdx.x * 256 + threadIdx.x;     // B*N*C = 2^22
  int c = idx & 255;
  int n = (idx >> 8) & 4095;
  int b = idx >> 20;
  xt[idx] = (bf16)x[((size_t)b * Cc + c) * Nn + n];
}

// q_w [Cout][Cin][3][3] -> qwt [Cout][tap=9][Cin-permuted] bf16 (A-packed)
__global__ __launch_bounds__(256) void prep_qw(const float* __restrict__ qw,
                                               bf16* __restrict__ qwt) {
  int idx = blockIdx.x * 256 + threadIdx.x;     // 256*9*256 = 589824
  int ci_pos = idx & 255;
  int rest = idx >> 8;
  int t = rest % 9;
  int co = rest / 9;
  int ci_src = (ci_pos & ~31) | swap34(ci_pos & 31);
  qwt[idx] = (bf16)qw[((size_t)co * Cc + ci_src) * 9 + t];
}

// k_w/v_w [Cout][Cin] -> A-packed bf16; also zero the OOB guard buffer.
__global__ __launch_bounds__(256) void prep_kv(const float* __restrict__ kw,
                                               const float* __restrict__ vw,
                                               bf16* __restrict__ kwb,
                                               bf16* __restrict__ vwb,
                                               bf16* __restrict__ zguard) {
  int idx = blockIdx.x * 256 + threadIdx.x;     // 256*256 = 65536
  int src = (idx & ~31) | swap34(idx & 31);
  kwb[idx] = (bf16)kw[src];
  vwb[idx] = (bf16)vw[src];
  if (idx < 32) zguard[idx] = (bf16)0.f;
}

// ---------------------------------------------------------------------------
// qkv: implicit-GEMM conv via bf16 WMMA, one wave = 32(co) x 16(pixel) tile.
// B-fragment (x) reused across 2 co-subtiles; A-fragment reused across the
// 3 dilations. OOB taps load from a zeroed guard buffer (no EXEC juggling).
//   q -> qt [B][N][C-permuted] bf16 (A-packed for attention)
//   k -> kt [B][N][C] bf16 (linear: attention B-operand)
//   v -> vn [B][C][N] bf16 (channel-major: PV B-operand)
// ---------------------------------------------------------------------------
__global__ __launch_bounds__(128) void qkv_kernel(
    const bf16* __restrict__ xt, const bf16* __restrict__ qwt,
    const bf16* __restrict__ kwb, const bf16* __restrict__ vwb,
    const float* __restrict__ qb, const float* __restrict__ kb,
    const float* __restrict__ vb, const bf16* __restrict__ zguard,
    bf16* __restrict__ qt, bf16* __restrict__ kt, bf16* __restrict__ vn) {
  const int lane = threadIdx.x & 31;
  const int wid  = blockIdx.x * 4 + (threadIdx.x >> 5);   // 8192 wave-tiles
  const int nt = wid & 255;          // n-tile (16 px)
  const int ct = (wid >> 8) & 7;     // co-tile (32 ch)
  const int b  = wid >> 11;
  const int nbase = nt * 16, cobase = ct * 32;

  const int col = lane & 15;
  const int lh  = lane >> 4;
  const int ao  = lh * 16;           // A-fragment offset in packed layout
  const int bko = lh * 16;           // B-fragment K offset (linear layout)

  const int n = nbase + col;
  const int y = n >> 6, xc = n & 63;

  const bf16* xb   = xt + (size_t)b * Nn * Cc;
  const bf16* xrow = xb + (size_t)n * Cc;
  const int coa0 = cobase + col;          // A row, co-subtile 0
  const int coa1 = cobase + 16 + col;     // A row, co-subtile 1

  // ---- q: sum over dilations {1,3,5} of SiLU(conv3x3_dilated) --------------
  v8f qacc[2][3];
  #pragma unroll
  for (int s = 0; s < 2; ++s)
    #pragma unroll
    for (int d = 0; d < 3; ++d) qacc[s][d] = (v8f){};

  for (int cib = 0; cib < Cc; cib += 32) {
    #pragma unroll
    for (int t = 0; t < 9; ++t) {
      v16bf A0 = *(const v16bf*)(qwt + ((size_t)coa0 * 9 + t) * Cc + cib + ao);
      v16bf A1 = *(const v16bf*)(qwt + ((size_t)coa1 * 9 + t) * Cc + cib + ao);
      const int ky = t / 3 - 1, kx = t % 3 - 1;
      #pragma unroll
      for (int di = 0; di < 3; ++di) {
        const int dil = (di == 0) ? 1 : (di == 1) ? 3 : 5;
        const int sy = y + ky * dil, sx = xc + kx * dil;
        const bool ok = ((unsigned)sy < 64u) & ((unsigned)sx < 64u);
        const bf16* bsrc =
            ok ? (xb + (size_t)(sy * 64 + sx) * Cc + cib + bko) : zguard;
        v16bf bfrag = *(const v16bf*)bsrc;
        qacc[0][di] = WMMA_BF16(A0, bfrag, qacc[0][di]);
        qacc[1][di] = WMMA_BF16(A1, bfrag, qacc[1][di]);
      }
    }
  }
  #pragma unroll
  for (int s = 0; s < 2; ++s) {
    #pragma unroll
    for (int r = 0; r < 8; ++r) {
      const int j  = s * 16 + lh * 8 + r;      // co offset within 32-tile
      const int co = cobase + j;
      const float bias = qb[co];
      float z0 = qacc[s][0][r] + bias;
      float z1 = qacc[s][1][r] + bias;
      float z2 = qacc[s][2][r] + bias;
      float qv = z0 / (1.f + __expf(-z0)) + z1 / (1.f + __expf(-z1)) +
                 z2 / (1.f + __expf(-z2));
      // A-packed store for the attention q operand
      qt[((size_t)b * Nn + n) * Cc + cobase + swap34(j)] = (bf16)qv;
    }
  }

  // ---- k, v: 1x1 convs (one bx fragment feeds 4 WMMAs) ---------------------
  v8f kacc[2] = {(v8f){}, (v8f){}}, vacc[2] = {(v8f){}, (v8f){}};
  for (int cib = 0; cib < Cc; cib += 32) {
    v16bf bx = *(const v16bf*)(xrow + cib + bko);
    #pragma unroll
    for (int s = 0; s < 2; ++s) {
      const int coa = cobase + s * 16 + col;
      v16bf AK = *(const v16bf*)(kwb + (size_t)coa * Cc + cib + ao);
      v16bf AV = *(const v16bf*)(vwb + (size_t)coa * Cc + cib + ao);
      kacc[s] = WMMA_BF16(AK, bx, kacc[s]);
      vacc[s] = WMMA_BF16(AV, bx, vacc[s]);
    }
  }
  #pragma unroll
  for (int s = 0; s < 2; ++s) {
    #pragma unroll
    for (int r = 0; r < 8; ++r) {
      const int co = cobase + s * 16 + lh * 8 + r;
      kt[((size_t)b * Nn + n) * Cc + co] = (bf16)(kacc[s][r] + kb[co]);
      vn[((size_t)b * Cc + co) * Nn + n] = (bf16)(vacc[s][r] + vb[co]);
    }
  }
}

// ---------------------------------------------------------------------------
// Flash-style attention, key-block Bc=64, with LDS-staged k/v shared by the
// 4 waves of a block (same (b,h), consecutive 16-row query tiles).
// Per step and wave: 16 WMMAs S = q^T k (operands from LDS), one online
// softmax pass (exp2 domain), P staged via 2 KB/wave LDS tile (A-packed),
// 16 WMMAs O += P * v^T (operands from LDS). k/v tiles (64x128 + 128x64 bf16,
// 32 KB) are double-buffered; staging overlaps the PV phase. This cuts k/v
// global/L2 traffic 4x vs per-wave streaming.
// ---------------------------------------------------------------------------
__global__ __launch_bounds__(128) void attn_kernel(
    const bf16* __restrict__ qt, const bf16* __restrict__ kt,
    const bf16* __restrict__ vn, float* __restrict__ out) {
  __shared__ bf16 kbuf[2][64 * 128];   // [m within step][d]   16 KB each
  __shared__ bf16 vbuf[2][128 * 64];   // [d][m within step]   16 KB each
  __shared__ bf16 Plds[4][16 * 64];    // per-wave P tile       8 KB total

  const int tid  = threadIdx.x;
  const int lane = tid & 31;
  const int wv   = tid >> 5;
  const int ntg  = blockIdx.x & 63;          // 64 query-groups of 64 rows
  const int h    = (blockIdx.x >> 6) & 1;
  const int b    = blockIdx.x >> 7;
  const int nbase = ntg * 64 + wv * 16;      // this wave's 16 query rows

  const int col = lane & 15;
  const int lh  = lane >> 4;
  const int ao  = lh * 16;
  const int bko = lh * 16;
  // (1/sqrt(128)) * log2(e): softmax in exp2 domain
  const float escale = 0.0883883476483184405f * 1.4426950408889634f;

  const bf16* kb_ = kt + (size_t)b * Nn * Cc + h * HD;   // [m][128] rows, stride Cc
  const bf16* vb_ = vn + ((size_t)b * Cc + h * HD) * Nn; // [d][Nn] rows

  // cooperative stage of one 64-key step into buffer `buf`
  auto stage = [&](int buf, int mbs) {
    #pragma unroll
    for (int i = 0; i < 8; ++i) {            // k tile: 1024 x 16B vectors
      int v = i * 128 + tid;
      int row = v >> 4, off = (v & 15) * 8;
      *(uint4*)(&kbuf[buf][row * 128 + off]) =
          *(const uint4*)(kb_ + (size_t)(mbs + row) * Cc + off);
    }
    #pragma unroll
    for (int i = 0; i < 8; ++i) {            // v tile: 1024 x 16B vectors
      int v = i * 128 + tid;
      int row = v >> 3, off = (v & 7) * 8;
      *(uint4*)(&vbuf[buf][row * 64 + off]) =
          *(const uint4*)(vb_ + (size_t)row * Nn + mbs + off);
    }
  };

  // q A-fragments: qt is A-packed => one v16bf load per 32-d chunk
  const bf16* qrow = qt + ((size_t)b * Nn + (nbase + col)) * Cc + h * HD;
  v16bf aq[4];
  #pragma unroll
  for (int db = 0; db < 4; ++db)
    aq[db] = *(const v16bf*)(qrow + db * 32 + ao);

  v8f O[8];
  #pragma unroll
  for (int i = 0; i < 8; ++i) O[i] = (v8f){};
  float mrow[8], lrow[8];
  #pragma unroll
  for (int r = 0; r < 8; ++r) { mrow[r] = -3.0e38f; lrow[r] = 0.f; }

  // linear positions (within a 32-col chunk) of this lane's P values
  const int pos0 = col;        // chunk offset col
  const int pos1 = col + 16;   // chunk offset 16+col

  stage(0, 0);
  __syncthreads();

  for (int mb = 0; mb < Nn; mb += 64) {
    const int cur = (mb >> 6) & 1;

    // ---- S tile [16 n x 64 m] from LDS k-tile ----
    v8f s[4];
    #pragma unroll
    for (int ms = 0; ms < 4; ++ms) s[ms] = (v8f){};
    #pragma unroll
    for (int db = 0; db < 4; ++db) {
      #pragma unroll
      for (int ms = 0; ms < 4; ++ms) {
        v16bf bk =
            *(const v16bf*)(&kbuf[cur][(ms * 16 + col) * 128 + db * 32 + bko]);
        s[ms] = WMMA_BF16(aq[db], bk, s[ms]);
      }
    }

    __syncthreads();  // WAR on Plds vs previous iteration's reads
    // ---- online softmax (row = r + 8*lh lives in one 16-lane half) ----
    #pragma unroll
    for (int r = 0; r < 8; ++r) {
      float e0 = s[0][r] * escale, e1 = s[1][r] * escale;
      float e2 = s[2][r] * escale, e3 = s[3][r] * escale;
      float vmax = fmaxf(fmaxf(e0, e1), fmaxf(e2, e3));
      #pragma unroll
      for (int off = 1; off < 16; off <<= 1)
        vmax = fmaxf(vmax, __shfl_xor(vmax, off, 32));
      float mnew  = fmaxf(mrow[r], vmax);
      float alpha = exp2f(mrow[r] - mnew);
      float p0 = exp2f(e0 - mnew), p1 = exp2f(e1 - mnew);
      float p2 = exp2f(e2 - mnew), p3 = exp2f(e3 - mnew);
      float rs = (p0 + p1) + (p2 + p3);
      #pragma unroll
      for (int off = 1; off < 16; off <<= 1)
        rs += __shfl_xor(rs, off, 32);
      lrow[r] = lrow[r] * alpha + rs;
      mrow[r] = mnew;
      #pragma unroll
      for (int db = 0; db < 8; ++db) O[db][r] *= alpha;
      const int row = r + lh * 8;
      bf16* prow = &Plds[wv][row * 64];
      prow[pos0]      = (bf16)p0;   // m = col        (chunk 0)
      prow[pos1]      = (bf16)p1;   // m = 16 + col   (chunk 0)
      prow[32 + pos0] = (bf16)p2;   // m = 32 + col   (chunk 1)
      prow[32 + pos1] = (bf16)p3;   // m = 48 + col   (chunk 1)
    }
    __syncthreads();  // publish P before A-fragment reads

    // ---- stage next k/v step: overlaps the PV phase ----
    if (mb + 64 < Nn) stage(cur ^ 1, mb + 64);

    // ---- P A-fragments: one contiguous v16bf per 32-col K-chunk ----
    v16bf aP0 = *(const v16bf*)(&Plds[wv][col * 64 + ao]);
    v16bf aP1 = *(const v16bf*)(&Plds[wv][col * 64 + 32 + ao]);

    // ---- O += P [16n x 64m] * v^T [64m x 16d] from LDS v-tile ----
    #pragma unroll
    for (int db = 0; db < 8; ++db) {
      const bf16* vrow = &vbuf[cur][(db * 16 + col) * 64 + bko];
      v16bf bv0 = *(const v16bf*)(vrow);
      v16bf bv1 = *(const v16bf*)(vrow + 32);
      O[db] = WMMA_BF16(aP0, bv0, O[db]);
      O[db] = WMMA_BF16(aP1, bv1, O[db]);
    }

    __syncthreads();  // cur fully read; next buffer published
  }

  // ---- epilogue: out[b, h*128+d, n] = O[n][d] / lrow ----
  float* ob = out + ((size_t)b * Cc + h * HD) * Nn;
  #pragma unroll
  for (int db = 0; db < 8; ++db) {
    const int d = db * 16 + col;
    #pragma unroll
    for (int r = 0; r < 8; ++r) {
      const int nn = nbase + r + lh * 8;
      ob[(size_t)d * Nn + nn] = O[db][r] / lrow[r];
    }
  }
}

// ---------------------------------------------------------------------------
extern "C" void kernel_launch(void* const* d_in, const int* in_sizes, int n_in,
                              void* d_out, int out_size, void* d_ws, size_t ws_size,
                              hipStream_t stream) {
  (void)in_sizes; (void)n_in; (void)out_size; (void)ws_size;
  const float* x  = (const float*)d_in[0];
  const float* qw = (const float*)d_in[1];
  const float* qb = (const float*)d_in[2];
  const float* kw = (const float*)d_in[3];
  const float* kb = (const float*)d_in[4];
  const float* vw = (const float*)d_in[5];
  const float* vb = (const float*)d_in[6];
  float* out = (float*)d_out;

  char* ws = (char*)d_ws;
  const size_t SZT = (size_t)Bn * Nn * Cc * sizeof(bf16);   // 8 MiB per tensor
  bf16* xt  = (bf16*)(ws);
  bf16* qt  = (bf16*)(ws + SZT);
  bf16* kt  = (bf16*)(ws + 2 * SZT);
  bf16* vn  = (bf16*)(ws + 3 * SZT);
  bf16* qwt = (bf16*)(ws + 4 * SZT);
  bf16* kwb = (bf16*)(ws + 4 * SZT + (size_t)Cc * 9 * Cc * sizeof(bf16));
  bf16* vwb = (bf16*)(ws + 4 * SZT + (size_t)Cc * 9 * Cc * sizeof(bf16)
                         + (size_t)Cc * Cc * sizeof(bf16));
  bf16* zg  = (bf16*)(ws + 4 * SZT + (size_t)Cc * 9 * Cc * sizeof(bf16)
                         + 2 * (size_t)Cc * Cc * sizeof(bf16));

  prep_x <<<16384, 256, 0, stream>>>(x, xt);
  prep_qw<<< 2304, 256, 0, stream>>>(qw, qwt);
  prep_kv<<<  256, 256, 0, stream>>>(kw, vw, kwb, vwb, zg);
  qkv_kernel<<<2048, 128, 0, stream>>>(xt, qwt, kwb, vwb, qb, kb, vb, zg,
                                       qt, kt, vn);
  attn_kernel<<<512, 128, 0, stream>>>(qt, kt, vn, out);
}
